// DynamicWeightNetwork_23905787969971
// MI455X (gfx1250) — matrix operand order repaired and checked by reading, not compiled, not run on hardware
//
#include <hip/hip_runtime.h>

typedef __attribute__((ext_vector_type(8)))  float   v8f;
typedef __attribute__((ext_vector_type(16))) __bf16  v16bf;
typedef __attribute__((ext_vector_type(8)))  __bf16  bf16x8;

#define NB   4096   // batch
#define NIN  512
#define NE   64     // experts = T*C
#define NOUTL 256   // final layer width
#define H1W  896    // 128 (classifier L1) + 12*64 (selector L1s)
#define ROWP 40     // padded LDS row stride (elements) -> conflict-free ds_load_b128
#define ASZ  (64 * ROWP)
#define BSZ  (128 * ROWP)

static __device__ __forceinline__ v16bf mk16(bf16x8 lo, bf16x8 hi) {
  v16bf r;
#pragma unroll
  for (int i = 0; i < 8; ++i) { r[i] = lo[i]; r[i + 8] = hi[i]; }
  return r;
}

static __device__ __forceinline__ bf16x8 cvt8(float4 a, float4 b, float mv) {
  bf16x8 r;
  r[0] = (__bf16)(a.x * mv); r[1] = (__bf16)(a.y * mv);
  r[2] = (__bf16)(a.z * mv); r[3] = (__bf16)(a.w * mv);
  r[4] = (__bf16)(b.x * mv); r[5] = (__bf16)(b.y * mv);
  r[6] = (__bf16)(b.z * mv); r[7] = (__bf16)(b.w * mv);
  return r;
}

// wave-level 32x64 WMMA tile update from one KT=32 LDS stage
static __device__ __forceinline__ void wave_mma(const __bf16* As, const __bf16* Bs,
                                                int wr, int wc, int lm, int lh,
                                                v8f acc[2][4]) {
  const __bf16* Aw = As + (wr * 32) * ROWP;
  const __bf16* Bw = Bs + (wc * 64) * ROWP;
  v16bf afr[2], bfr[4];
#pragma unroll
  for (int mt = 0; mt < 2; ++mt) {
    const __bf16* p = Aw + (mt * 16 + lm) * ROWP;
    afr[mt] = mk16(*(const bf16x8*)(p + lh * 8), *(const bf16x8*)(p + 16 + lh * 8));
  }
#pragma unroll
  for (int nt = 0; nt < 4; ++nt) {
    const __bf16* q = Bw + (nt * 16 + lm) * ROWP + lh * 8;
    bfr[nt] = mk16(*(const bf16x8*)(q), *(const bf16x8*)(q + 16));
  }
#pragma unroll
  for (int mt = 0; mt < 2; ++mt)
#pragma unroll
    for (int nt = 0; nt < 4; ++nt)
      acc[mt][nt] = __builtin_amdgcn_wmma_f32_16x16x32_bf16(
          false, afr[mt], false, bfr[nt], (short)0, acc[mt][nt], false, false);
}

// ---------------- conversion / copy helpers ----------------
__global__ void k_cvt_bf16(const float* __restrict__ s, __bf16* __restrict__ d, int n) {
  for (int i = blockIdx.x * blockDim.x + threadIdx.x; i < n; i += gridDim.x * blockDim.x)
    d[i] = (__bf16)s[i];
}
__global__ void k_copy_f32(const float* __restrict__ s, float* __restrict__ d, int n) {
  int i = blockIdx.x * blockDim.x + threadIdx.x;
  if (i < n) d[i] = s[i];
}

// ---------------- generic bf16 GEMM: out = relu(A[M,K] @ W[N,K]^T + bias) ----------------
// block = 128 threads (4 waves), block tile 64(M) x 128(N), KT=32, double-buffered LDS
__global__ void __launch_bounds__(128, 1)
k_lin_gemm_relu(const __bf16* __restrict__ A, const __bf16* __restrict__ W,
                const float* __restrict__ bias, float* __restrict__ out,
                int K, int N) {
  __shared__ __align__(16) __bf16 As[2 * ASZ];
  __shared__ __align__(16) __bf16 Bs[2 * BSZ];
  const int tid = threadIdx.x;
  const int b0 = blockIdx.x * 64, n0 = blockIdx.y * 128;
  const int ar = tid >> 1, ah = tid & 1;
  const int lane = tid & 31, wv = tid >> 5;
  const int wr = wv >> 1, wc = wv & 1, lm = lane & 15, lh = lane >> 4;

  const __bf16* Arow = A + (size_t)(b0 + ar) * K + ah * 16;
  const __bf16* Wrow = W + (size_t)(n0 + tid) * K;

  v8f acc[2][4];
#pragma unroll
  for (int mt = 0; mt < 2; ++mt)
#pragma unroll
    for (int nt = 0; nt < 4; ++nt)
#pragma unroll
      for (int i = 0; i < 8; ++i) acc[mt][nt][i] = 0.0f;

  const int total = K >> 5;

  // prologue: stage iteration 0 into buffer 0
  bf16x8 a0, a1, w0, w1, w2, w3;
  {
    const bf16x8* ap = (const bf16x8*)(Arow);
    a0 = ap[0]; a1 = ap[1];
    const bf16x8* wp = (const bf16x8*)(Wrow);
    w0 = wp[0]; w1 = wp[1]; w2 = wp[2]; w3 = wp[3];
    *(bf16x8*)&As[ar * ROWP + ah * 16]     = a0;
    *(bf16x8*)&As[ar * ROWP + ah * 16 + 8] = a1;
    *(bf16x8*)&Bs[tid * ROWP]      = w0;
    *(bf16x8*)&Bs[tid * ROWP + 8]  = w1;
    *(bf16x8*)&Bs[tid * ROWP + 16] = w2;
    *(bf16x8*)&Bs[tid * ROWP + 24] = w3;
  }
  __syncthreads();

  int buf = 0;
  for (int it = 0; it < total - 1; ++it) {   // steady state, branch-free body
    const int kb = (it + 1) << 5;
    const bf16x8* ap = (const bf16x8*)(Arow + kb);
    a0 = ap[0]; a1 = ap[1];
    const bf16x8* wp = (const bf16x8*)(Wrow + kb);
    w0 = wp[0]; w1 = wp[1]; w2 = wp[2]; w3 = wp[3];

    wave_mma(As + buf * ASZ, Bs + buf * BSZ, wr, wc, lm, lh, acc);

    const int nb = buf ^ 1;
    *(bf16x8*)&As[nb * ASZ + ar * ROWP + ah * 16]     = a0;
    *(bf16x8*)&As[nb * ASZ + ar * ROWP + ah * 16 + 8] = a1;
    *(bf16x8*)&Bs[nb * BSZ + tid * ROWP]      = w0;
    *(bf16x8*)&Bs[nb * BSZ + tid * ROWP + 8]  = w1;
    *(bf16x8*)&Bs[nb * BSZ + tid * ROWP + 16] = w2;
    *(bf16x8*)&Bs[nb * BSZ + tid * ROWP + 24] = w3;
    __syncthreads();
    buf = nb;
  }
  wave_mma(As + buf * ASZ, Bs + buf * BSZ, wr, wc, lm, lh, acc);  // epilogue compute

#pragma unroll
  for (int mt = 0; mt < 2; ++mt)
#pragma unroll
    for (int nt = 0; nt < 4; ++nt) {
      const int col = n0 + wc * 64 + nt * 16 + lm;
#pragma unroll
      for (int r = 0; r < 8; ++r) {
        const int row = b0 + wr * 32 + mt * 16 + r + 8 * lh;
        float v = acc[mt][nt][r] + bias[col];
        out[(size_t)row * N + col] = fmaxf(v, 0.0f);
      }
    }
}

// ---------------- fused MoE GEMM: out = act( sum_e mix[b,e]*(cur[b,:]@W[e]^T) + bmix[b,:] ) ----
// cur fp32 [B,512]; W bf16 [64][N][512]; mix fp32 [B,64]; bmix fp32 [B,N]
// Effective K = 64*512 = 32768, expert scale folded into A staging; double-buffered pipeline.
__global__ void __launch_bounds__(128, 1)
k_moe_gemm(const float* __restrict__ cur, const __bf16* __restrict__ W,
           const float* __restrict__ mix, const float* __restrict__ bmix,
           float* __restrict__ out, int N, int relu) {
  __shared__ __align__(16) __bf16 As[2 * ASZ];
  __shared__ __align__(16) __bf16 Bs[2 * BSZ];
  const int tid = threadIdx.x;
  const int b0 = blockIdx.x * 64, n0 = blockIdx.y * 128;
  const int ar = tid >> 1, ah = tid & 1;
  const int lane = tid & 31, wv = tid >> 5;
  const int wr = wv >> 1, wc = wv & 1, lm = lane & 15, lh = lane >> 4;

  const float* Arow = cur + (size_t)(b0 + ar) * NIN + ah * 16;
  const float* mrow = mix + (size_t)(b0 + ar) * NE;

  v8f acc[2][4];
#pragma unroll
  for (int mt = 0; mt < 2; ++mt)
#pragma unroll
    for (int nt = 0; nt < 4; ++nt)
#pragma unroll
      for (int i = 0; i < 8; ++i) acc[mt][nt][i] = 0.0f;

  const int total = NE * (NIN >> 5);   // 1024 K-steps of 32

  float4 x0, x1, x2, x3;               // explicit regs (no indexable array -> no scratch)
  float mv;
  bf16x8 w0, w1, w2, w3;

  // prologue: global loads + stage into buffer 0 (it = 0 -> e = 0, kb = 0)
  {
    mv = mrow[0];
    const float4* ap = (const float4*)(Arow);
    x0 = ap[0]; x1 = ap[1]; x2 = ap[2]; x3 = ap[3];
    const bf16x8* wp = (const bf16x8*)(W + (size_t)(n0 + tid) * NIN);
    w0 = wp[0]; w1 = wp[1]; w2 = wp[2]; w3 = wp[3];
    *(bf16x8*)&As[ar * ROWP + ah * 16]     = cvt8(x0, x1, mv);
    *(bf16x8*)&As[ar * ROWP + ah * 16 + 8] = cvt8(x2, x3, mv);
    *(bf16x8*)&Bs[tid * ROWP]      = w0;
    *(bf16x8*)&Bs[tid * ROWP + 8]  = w1;
    *(bf16x8*)&Bs[tid * ROWP + 16] = w2;
    *(bf16x8*)&Bs[tid * ROWP + 24] = w3;
  }
  __syncthreads();

  int buf = 0;
  for (int it = 0; it < total - 1; ++it) {   // steady state, branch-free body
    const int nit = it + 1;
    const int e  = nit >> 4;
    const int kb = (nit & 15) << 5;
    mv = mrow[e];
    const float4* ap = (const float4*)(Arow + kb);
    x0 = ap[0]; x1 = ap[1]; x2 = ap[2]; x3 = ap[3];
    const bf16x8* wp = (const bf16x8*)(W + ((size_t)e * N + n0 + tid) * NIN + kb);
    w0 = wp[0]; w1 = wp[1]; w2 = wp[2]; w3 = wp[3];

    // prefetch B stream a few K-steps ahead (lowers to global_prefetch_b8)
    const int pit = it + 4;
    if (pit < total) {
      const int pe = pit >> 4;
      const int pkb = (pit & 15) << 5;
      __builtin_prefetch(W + ((size_t)pe * N + n0 + tid) * NIN + pkb, 0, 3);
    }

    wave_mma(As + buf * ASZ, Bs + buf * BSZ, wr, wc, lm, lh, acc);

    const int nb = buf ^ 1;
    *(bf16x8*)&As[nb * ASZ + ar * ROWP + ah * 16]     = cvt8(x0, x1, mv);
    *(bf16x8*)&As[nb * ASZ + ar * ROWP + ah * 16 + 8] = cvt8(x2, x3, mv);
    *(bf16x8*)&Bs[nb * BSZ + tid * ROWP]      = w0;
    *(bf16x8*)&Bs[nb * BSZ + tid * ROWP + 8]  = w1;
    *(bf16x8*)&Bs[nb * BSZ + tid * ROWP + 16] = w2;
    *(bf16x8*)&Bs[nb * BSZ + tid * ROWP + 24] = w3;
    __syncthreads();
    buf = nb;
  }
  wave_mma(As + buf * ASZ, Bs + buf * BSZ, wr, wc, lm, lh, acc);  // epilogue compute

#pragma unroll
  for (int mt = 0; mt < 2; ++mt)
#pragma unroll
    for (int nt = 0; nt < 4; ++nt) {
      const int col = n0 + wc * 64 + nt * 16 + lm;
#pragma unroll
      for (int r = 0; r < 8; ++r) {
        const int row = b0 + wr * 32 + mt * 16 + r + 8 * lh;
        float v = acc[mt][nt][r] + bmix[(size_t)row * N + col];
        if (relu) v = fmaxf(v, 0.0f);
        out[(size_t)row * N + col] = v;
      }
    }
}

// ---------------- classifier tail: h1[0:128] -> 64 -> 32 -> 4 -> softmax ----------------
__global__ void k_classifier_tail(const float* __restrict__ h1,
                                  const float* Wc2, const float* bc2,
                                  const float* Wc3, const float* bc3,
                                  const float* Wc4, const float* bc4,
                                  float* __restrict__ tp) {
  const int b = blockIdx.x, t = threadIdx.x;
  __shared__ float sh[128];
  sh[t] = h1[(size_t)b * H1W + t];   // already ReLU'd by lin_gemm
  __syncthreads();
  float v = 0.0f;
  if (t < 64) {
    v = bc2[t];
    for (int k = 0; k < 128; ++k) v += Wc2[t * 128 + k] * sh[k];
    v = fmaxf(v, 0.0f);
  }
  __syncthreads(); if (t < 64) sh[t] = v; __syncthreads();
  if (t < 32) {
    v = bc3[t];
    for (int k = 0; k < 64; ++k) v += Wc3[t * 64 + k] * sh[k];
    v = fmaxf(v, 0.0f);
  }
  __syncthreads(); if (t < 32) sh[t] = v; __syncthreads();
  if (t < 4) {
    v = bc4[t];
    for (int k = 0; k < 32; ++k) v += Wc4[t * 32 + k] * sh[k];
  }
  __syncthreads(); if (t < 4) sh[t] = v; __syncthreads();
  if (t < 4) {
    float m = fmaxf(fmaxf(sh[0], sh[1]), fmaxf(sh[2], sh[3]));
    float s = expf(sh[0] - m) + expf(sh[1] - m) + expf(sh[2] - m) + expf(sh[3] - m);
    tp[b * 4 + t] = expf(sh[t] - m) / s;
  }
}

// ---------------- selector tail: s1 -> 32 -> 16 -> softmax, combine with task probs --------
__global__ void k_selector_tail(const float* __restrict__ h1,
                                const float* __restrict__ S2, const float* __restrict__ b2,
                                const float* __restrict__ S3, const float* __restrict__ b3,
                                const float* __restrict__ tp, float* __restrict__ mix) {
  const int b = blockIdx.x, lt = blockIdx.y, lane = threadIdx.x;
  const int l = lt >> 2, t = lt & 3;
  __shared__ float s1s[64], s2s[32], lg[16];
  const float* hrow = h1 + (size_t)b * H1W + 128 + lt * 64;  // already ReLU'd
  s1s[lane] = hrow[lane];
  s1s[lane + 32] = hrow[lane + 32];
  __syncthreads();
  float v = b2[lt * 32 + lane];
  const float* w2 = S2 + (size_t)(lt * 32 + lane) * 64;
  for (int k = 0; k < 64; ++k) v += w2[k] * s1s[k];
  s2s[lane] = fmaxf(v, 0.0f);
  __syncthreads();
  if (lane < 16) {
    float u = b3[lt * 16 + lane];
    const float* w3 = S3 + (size_t)(lt * 16 + lane) * 32;
    for (int k = 0; k < 32; ++k) u += w3[k] * s2s[k];
    lg[lane] = u;
  }
  __syncthreads();
  if (lane < 16) {
    float m = -1e30f;
    for (int i = 0; i < 16; ++i) m = fmaxf(m, lg[i]);
    float s = 0.0f;
    for (int i = 0; i < 16; ++i) s += expf(lg[i] - m);
    float p = expf(lg[lane] - m) / s;
    mix[((size_t)l * NB + b) * NE + t * 16 + lane] = tp[b * 4 + t] * p;
  }
}

// ---------------- pre-mixed bias: bmix[b,o] = sum_e mix[b,e]*bias[e,o] ----------------
__global__ void k_bias_mix(const float* __restrict__ mix, const float* __restrict__ bias,
                           float* __restrict__ bmix, int N) {
  const int b = blockIdx.x;
  const int o = blockIdx.y * 256 + threadIdx.x;
  __shared__ float mr[NE];
  if (threadIdx.x < NE) mr[threadIdx.x] = mix[(size_t)b * NE + threadIdx.x];
  __syncthreads();
  float acc = 0.0f;
  for (int e = 0; e < NE; ++e) acc += mr[e] * bias[(size_t)e * N + o];
  bmix[(size_t)b * N + o] = acc;
}

// ---------------- host launch ----------------
extern "C" void kernel_launch(void* const* d_in, const int* in_sizes, int n_in,
                              void* d_out, int out_size, void* d_ws, size_t ws_size,
                              hipStream_t stream) {
  const float* x   = (const float*)d_in[0];
  const float* Wc1 = (const float*)d_in[1];  const float* bc1 = (const float*)d_in[2];
  const float* Wc2 = (const float*)d_in[3];  const float* bc2 = (const float*)d_in[4];
  const float* Wc3 = (const float*)d_in[5];  const float* bc3 = (const float*)d_in[6];
  const float* Wc4 = (const float*)d_in[7];  const float* bc4 = (const float*)d_in[8];
  const float* S1  = (const float*)d_in[9];  const float* b1  = (const float*)d_in[10];
  const float* S2  = (const float*)d_in[11]; const float* b2  = (const float*)d_in[12];
  const float* S3  = (const float*)d_in[13]; const float* b3  = (const float*)d_in[14];
  const float* Wh  = (const float*)d_in[15]; const float* bh  = (const float*)d_in[16];
  const float* Wo  = (const float*)d_in[17]; const float* bo  = (const float*)d_in[18];

  char* ws = (char*)d_ws;
  size_t off = 0;
  auto take = [&](size_t bytes) -> void* {
    void* p = ws + off;
    off += (bytes + 255) & ~(size_t)255;
    return p;
  };
  __bf16* xbf   = (__bf16*)take((size_t)NB * NIN * 2);
  __bf16* W1cat = (__bf16*)take((size_t)H1W * NIN * 2);
  float*  b1cat = (float*)take((size_t)H1W * 4);
  __bf16* Whbf  = (__bf16*)take((size_t)2 * NE * NIN * NIN * 2);
  __bf16* Wobf  = (__bf16*)take((size_t)NE * NOUTL * NIN * 2);
  float*  h1    = (float*)take((size_t)NB * H1W * 4);
  float*  mixb  = (float*)take((size_t)3 * NB * NE * 4);
  float*  bmix0 = (float*)take((size_t)NB * NIN * 4);
  float*  bmix1 = (float*)take((size_t)NB * NIN * 4);
  float*  bmix2 = (float*)take((size_t)NB * NOUTL * 4);
  float*  hA    = (float*)take((size_t)NB * NIN * 4);
  float*  hB    = (float*)take((size_t)NB * NIN * 4);

  float* outp = (float*)d_out;                        // [B, 256]
  float* tp   = (float*)d_out + (size_t)NB * NOUTL;   // [B, 4] task_probs

  // --- convert to bf16 ---
  k_cvt_bf16<<<dim3(2048), 256, 0, stream>>>(x, xbf, NB * NIN);
  k_cvt_bf16<<<dim3(512), 256, 0, stream>>>(Wc1, W1cat, 128 * NIN);
  k_cvt_bf16<<<dim3(2048), 256, 0, stream>>>(S1, W1cat + 128 * NIN, 768 * NIN);
  k_cvt_bf16<<<dim3(8192), 256, 0, stream>>>(Wh, Whbf, 2 * NE * NIN * NIN);
  k_cvt_bf16<<<dim3(4096), 256, 0, stream>>>(Wo, Wobf, NE * NOUTL * NIN);
  k_copy_f32<<<dim3(1), 256, 0, stream>>>(bc1, b1cat, 128);
  k_copy_f32<<<dim3(3), 256, 0, stream>>>(b1, b1cat + 128, 768);

  // --- fused L1: classifier layer1 + all 12 selector layer1s (WMMA GEMM, ReLU) ---
  k_lin_gemm_relu<<<dim3(NB / 64, H1W / 128), 128, 0, stream>>>(xbf, W1cat, b1cat, h1, NIN, H1W);

  // --- small tails (VALU) ---
  k_classifier_tail<<<dim3(NB), 128, 0, stream>>>(h1, Wc2, bc2, Wc3, bc3, Wc4, bc4, tp);
  k_selector_tail<<<dim3(NB, 12), 32, 0, stream>>>(h1, S2, b2, S3, b3, tp, mixb);

  // --- pre-mixed biases ---
  k_bias_mix<<<dim3(NB, 2), 256, 0, stream>>>(mixb, bh, bmix0, NIN);
  k_bias_mix<<<dim3(NB, 2), 256, 0, stream>>>(mixb + (size_t)NB * NE, bh + (size_t)NE * NIN, bmix1, NIN);
  k_bias_mix<<<dim3(NB, 1), 256, 0, stream>>>(mixb + (size_t)2 * NB * NE, bo, bmix2, NOUTL);

  // --- fused MoE GEMMs (the 343 GFLOP) ---
  k_moe_gemm<<<dim3(NB / 64, NIN / 128), 128, 0, stream>>>(x, Whbf, mixb, bmix0, hA, NIN, 1);
  k_moe_gemm<<<dim3(NB / 64, NIN / 128), 128, 0, stream>>>(hA, Whbf + (size_t)NE * NIN * NIN,
                                                           mixb + (size_t)NB * NE, bmix1, hB, NIN, 1);
  k_moe_gemm<<<dim3(NB / 64, NOUTL / 128), 128, 0, stream>>>(hB, Wobf, mixb + (size_t)2 * NB * NE,
                                                             bmix2, outp, NOUTL, 0);

  (void)in_sizes; (void)n_in; (void)out_size; (void)ws_size;
}
